// EncoderBlock_66176856097005
// MI455X (gfx1250) — compile-verified, matrix operand order checked by reading, not run on hardware
//
#include <hip/hip_runtime.h>
#include <hip/hip_bf16.h>
#include <stdint.h>

// ---------------------------------------------------------------------------
// Transformer encoder block for MI455X (gfx1250, wave32, WMMA + TDM).
// B=8 S=1024 D=1024 H=16 DK=64 DFF=4096, fp32 in/out.
// Compute-bound (~223 GFLOP) -> all GEMMs + attention on
// v_wmma_f32_16x16x32_bf16 (bf16 A/B, f32 accum). Non-transposed LDS tiles
// (GEMM A, attention K) are fed by the Tensor Data Mover
// (tensor_load_to_lds, TENSORcnt) instead of VALU copy loops.
// ---------------------------------------------------------------------------

#define B_   8
#define S_   1024
#define D_   1024
#define H_   16
#define DK_  64
#define DFF_ 4096
#define EPS_ 1e-5f

typedef __attribute__((ext_vector_type(16))) __bf16 v16bf;
typedef __attribute__((ext_vector_type(8)))  __bf16 v8bf;
typedef __attribute__((ext_vector_type(8)))  float  v8f;
typedef __attribute__((ext_vector_type(4)))  unsigned int u32x4;
typedef __attribute__((ext_vector_type(8)))  int          i32x8;
typedef __attribute__((ext_vector_type(4)))  int          i32x4;

static __device__ __forceinline__ __bf16 f2bf(float f) {
  union { float f; uint32_t u; } v; v.f = f;
  uint32_t r = v.u + 0x7FFFu + ((v.u >> 16) & 1u);   // round-to-nearest-even
  uint16_t h = (uint16_t)(r >> 16);
  return __builtin_bit_cast(__bf16, h);
}

static __device__ __forceinline__ v16bf cat16(v8bf lo, v8bf hi) {
  return __builtin_shufflevector(lo, hi, 0,1,2,3,4,5,6,7,8,9,10,11,12,13,14,15);
}

// ---------------------------------------------------------------------------
// Tensor Data Mover: 2D tile (bf16) global -> LDS, per CDNA5 ISA §8 D# layout.
// Group0: count=1 | lds_addr | global_addr[56:0] | type=2.
// Group1: data_size=1 (2B) | tensor_dim0/1 | tile_dim0/1 | dim0 stride.
// Groups 2/3 (and the trailing group on this 6-arg toolchain) zero: 2D tensor.
// Issue is per-wave (EXEC ignored); descriptors are wave-uniform -> SGPRs.
// Completion via s_wait_tensorcnt.
// ---------------------------------------------------------------------------
static __device__ __forceinline__ void tdm_load_2d_bf16(
    const void* gaddr, uint32_t lds_off,
    uint32_t tile_d0, uint32_t tile_d1,
    uint32_t tensor_d0, uint32_t tensor_d1,
    uint64_t stride_d0_elems) {
  uint64_t ga = (uint64_t)(uintptr_t)gaddr;
  u32x4 g0;
  g0[0] = 1u;                                   // count=1, user descriptor
  g0[1] = lds_off;                              // lds_addr (bytes)
  g0[2] = (uint32_t)(ga & 0xFFFFFFFFu);         // global_addr[31:0]
  g0[3] = (uint32_t)((ga >> 32) & 0x1FFFFFFu)   // global_addr[56:32]
        | (2u << 30);                           // type = 2 ("image")
  i32x8 g1;
  g1[0] = (int)(1u << 16);                      // data_size=1 (2 bytes)
  g1[1] = (int)((tensor_d0 & 0xFFFFu) << 16);   // tensor_dim0[15:0]
  g1[2] = (int)(((tensor_d0 >> 16) & 0xFFFFu) | ((tensor_d1 & 0xFFFFu) << 16));
  g1[3] = (int)(((tensor_d1 >> 16) & 0xFFFFu) | ((tile_d0 & 0xFFFFu) << 16));
  g1[4] = (int)(tile_d1 & 0xFFFFu);             // tile_dim1; tile_dim2=0
  g1[5] = (int)(uint32_t)(stride_d0_elems & 0xFFFFFFFFu);
  g1[6] = (int)(uint32_t)((stride_d0_elems >> 32) & 0xFFFFu);
  g1[7] = 0;
  i32x4 z4; z4[0] = 0; z4[1] = 0; z4[2] = 0; z4[3] = 0;
  i32x8 z8;
#pragma unroll
  for (int i = 0; i < 8; ++i) z8[i] = 0;
  __builtin_amdgcn_tensor_load_to_lds(g0, g1, z4, z4, z8, 0);
}

// ---------------------------------------------------------------------------
__global__ void zero_stats_kernel(float* stats) {
  if (threadIdx.x < 32) stats[threadIdx.x] = 0.0f;
}

__global__ void cast_f32_bf16_kernel(const float* __restrict__ src,
                                     __bf16* __restrict__ dst, int n) {
  int i = blockIdx.x * blockDim.x + threadIdx.x;
  if (i < n) dst[i] = f2bf(src[i]);
}

// ---------------------------------------------------------------------------
// Tiled GEMM: C[M,N] = act(A[M,K] @ W[K,N] + bias).
// Block = 128 threads (4 waves). Block tile 64x64, K step 32.
// A tile (64x32, row-major, no transpose) is DMA'd by the TDM (wave 0 issues,
// waits TENSORcnt, then the workgroup barrier publishes it). B tile is loaded
// coalesced and scatter-stored transposed (Bs[n][k]) so both fragments load
// as two contiguous ds_load_b128 per lane (ISA 7.12.2 16-bit A layout:
// lane l holds row l%16, chunks at k = hi*8 and k = 16+hi*8, hi = l/16).
// ---------------------------------------------------------------------------
template<bool GELU, bool WF32, bool WBF16>
__launch_bounds__(128)
__global__ void gemm_wmma_kernel(const __bf16* __restrict__ A,
                                 const __bf16* __restrict__ W,
                                 const float* __restrict__ bias,
                                 float* __restrict__ Cf, __bf16* __restrict__ Cb,
                                 int M, int N, int K) {
  __shared__ __align__(16) __bf16 As[64][32];
  __shared__ __align__(16) __bf16 Bs[64][32];   // transposed: Bs[n][k]
  const int t    = threadIdx.x;
  const int w    = t >> 5;
  const int lane = t & 31;
  const int hi   = lane >> 4;
  const int lm   = lane & 15;
  const int m0   = blockIdx.y * 64;
  const int n0   = blockIdx.x * 64;
  const uint32_t as_off = (uint32_t)(uintptr_t)&As[0][0];  // LDS byte offset

  v8f acc[4];
#pragma unroll
  for (int j = 0; j < 4; ++j)
#pragma unroll
    for (int e = 0; e < 8; ++e) acc[j][e] = 0.0f;

  for (int kt = 0; kt < K; kt += 32) {
    // A tile via Tensor Data Mover: tile 32(K) x 64(M), row stride K.
    if (w == 0) {
      tdm_load_2d_bf16(&A[(size_t)m0 * K + kt], as_off,
                       /*tile*/32, 64, /*tensor*/32, 64, (uint64_t)K);
    }
    // Prefetch next W K-tile (global_prefetch_b8).
    if (kt + 32 < K) {
      __builtin_prefetch(&W[(size_t)(kt + 32 + (t >> 3)) * N + n0 + (t & 7) * 8], 0, 1);
    }
    // B tile: read W rows coalesced, scatter-store transposed into LDS.
#pragma unroll
    for (int i = 0; i < 2; ++i) {
      int c = t + i * 128;
      int kr = c >> 3, noff = (c & 7) * 8;
      v8bf wv = *(const v8bf*)&W[(size_t)(kt + kr) * N + n0 + noff];
#pragma unroll
      for (int e = 0; e < 8; ++e) Bs[noff + e][kr] = wv[e];
    }
    if (w == 0) __builtin_amdgcn_s_wait_tensorcnt(0);
    __syncthreads();

    const int m = w * 16 + lm;
    v16bf a = cat16(*(const v8bf*)&As[m][hi * 8],
                    *(const v8bf*)&As[m][16 + hi * 8]);
#pragma unroll
    for (int j = 0; j < 4; ++j) {
      const int n = j * 16 + lm;
      v16bf b = cat16(*(const v8bf*)&Bs[n][hi * 8],
                      *(const v8bf*)&Bs[n][16 + hi * 8]);
      acc[j] = __builtin_amdgcn_wmma_f32_16x16x32_bf16(
          false, a, false, b, (short)0, acc[j], false, false);
    }
    __syncthreads();
  }

  // Epilogue: C layout (ISA 7.12.2): element r of v8f is row hi*8+r, col lm.
#pragma unroll
  for (int j = 0; j < 4; ++j) {
    const int ng = n0 + j * 16 + lm;
    const float bval = bias[ng];
#pragma unroll
    for (int r = 0; r < 8; ++r) {
      const int mg = m0 + w * 16 + hi * 8 + r;
      float v = acc[j][r] + bval;
      if (GELU) v = 0.5f * v * (1.0f + erff(v * 0.70710678118f));
      if (WF32)  Cf[(size_t)mg * N + ng] = v;
      if (WBF16) Cb[(size_t)mg * N + ng] = f2bf(v);
    }
  }
}

// ---------------------------------------------------------------------------
// Flash attention: block = 4 waves, each wave one 16-row q tile (block covers
// 64 q rows of one (b,h)). K tile (row-major copy) is DMA'd by the TDM;
// V tile is transposed into LDS by VALU so V fragments load contiguous.
// Online softmax; P goes through per-wave LDS to convert C-layout ->
// A-fragment layout (DS ops are in-order per wave).
// ---------------------------------------------------------------------------
__launch_bounds__(128)
__global__ void attn_wmma_kernel(const __bf16* __restrict__ Q,
                                 const __bf16* __restrict__ K,
                                 const __bf16* __restrict__ V,
                                 float* __restrict__ ctx) {
  __shared__ __align__(16) __bf16 Kt[32][64];      // [k-row][dk]
  __shared__ __align__(16) __bf16 Vt[64][32];      // transposed: [dk][k-row]
  __shared__ __align__(16) __bf16 Pb[4][16][32];   // per-wave P staging
  const int t    = threadIdx.x;
  const int w    = t >> 5;
  const int lane = t & 31;
  const int hi   = lane >> 4;
  const int lm   = lane & 15;
  const int bh   = blockIdx.y;
  const int b    = bh / H_;
  const int h    = bh % H_;
  const int q0   = blockIdx.x * 64 + w * 16;
  const uint32_t kt_off = (uint32_t)(uintptr_t)&Kt[0][0];

  // Q fragments: rows contiguous along DK in memory -> two b128 per fragment.
  const size_t qbase = ((size_t)(b * S_ + q0 + lm) * H_ + h) * DK_;
  v16bf qa[2];
#pragma unroll
  for (int c = 0; c < 2; ++c)
    qa[c] = cat16(*(const v8bf*)&Q[qbase + c * 32 + hi * 8],
                  *(const v8bf*)&Q[qbase + c * 32 + 16 + hi * 8]);

  float mstate[8], lstate[8];
  v8f acc[4];
#pragma unroll
  for (int r = 0; r < 8; ++r) { mstate[r] = -1e30f; lstate[r] = 0.0f; }
#pragma unroll
  for (int j = 0; j < 4; ++j)
#pragma unroll
    for (int e = 0; e < 8; ++e) acc[j][e] = 0.0f;

  for (int kb0 = 0; kb0 < S_; kb0 += 32) {
    // K tile via TDM: tile 64(DK) x 32(k rows), row stride H*DK.
    if (w == 0) {
      tdm_load_2d_bf16(&K[((size_t)(b * S_ + kb0) * H_ + h) * DK_], kt_off,
                       /*tile*/64, 32, /*tensor*/64, 32, (uint64_t)(H_ * DK_));
    }
    // V tile: coalesced loads, transposed scatter-store.
#pragma unroll
    for (int i = 0; i < 2; ++i) {
      int c = t + i * 128;
      int kr = c >> 3, coff = (c & 7) * 8;
      size_t g = ((size_t)(b * S_ + kb0 + kr) * H_ + h) * DK_ + coff;
      v8bf vv = *(const v8bf*)&V[g];
#pragma unroll
      for (int e = 0; e < 8; ++e) Vt[coff + e][kr] = vv[e];
    }
    if (w == 0) __builtin_amdgcn_s_wait_tensorcnt(0);
    __syncthreads();

    // Scores: two 16-col tiles, each chained over DK (2 x WMMA K=32).
    float sc[2][8], rm[8];
#pragma unroll
    for (int r = 0; r < 8; ++r) rm[r] = -1e30f;
#pragma unroll
    for (int nt = 0; nt < 2; ++nt) {
      const int krow = nt * 16 + lm;
      v8f s;
#pragma unroll
      for (int e = 0; e < 8; ++e) s[e] = 0.0f;
#pragma unroll
      for (int c = 0; c < 2; ++c) {
        v16bf kf = cat16(*(const v8bf*)&Kt[krow][c * 32 + hi * 8],
                         *(const v8bf*)&Kt[krow][c * 32 + 16 + hi * 8]);
        s = __builtin_amdgcn_wmma_f32_16x16x32_bf16(
            false, qa[c], false, kf, (short)0, s, false, false);
      }
#pragma unroll
      for (int r = 0; r < 8; ++r) {
        sc[nt][r] = s[r] * 0.125f;                 // 1/sqrt(DK)
        rm[r] = fmaxf(rm[r], sc[nt][r]);
      }
    }
    // Row-max across the 16-lane group (xor masks stay within the half-wave).
#pragma unroll
    for (int r = 0; r < 8; ++r)
#pragma unroll
      for (int off = 8; off; off >>= 1)
        rm[r] = fmaxf(rm[r], __shfl_xor(rm[r], off, 32));

    float alpha[8], p[2][8], rs[8];
#pragma unroll
    for (int r = 0; r < 8; ++r) {
      float mn = fmaxf(mstate[r], rm[r]);
      alpha[r] = __expf(mstate[r] - mn);
      mstate[r] = mn;
      rs[r] = 0.0f;
    }
#pragma unroll
    for (int nt = 0; nt < 2; ++nt)
#pragma unroll
      for (int r = 0; r < 8; ++r) {
        p[nt][r] = __expf(sc[nt][r] - mstate[r]);
        rs[r] += p[nt][r];
      }
#pragma unroll
    for (int r = 0; r < 8; ++r) {
#pragma unroll
      for (int off = 8; off; off >>= 1) rs[r] += __shfl_xor(rs[r], off, 32);
      lstate[r] = lstate[r] * alpha[r] + rs[r];
    }
#pragma unroll
    for (int j = 0; j < 4; ++j)
#pragma unroll
      for (int r = 0; r < 8; ++r) acc[j][r] *= alpha[r];

    // P: C-layout -> LDS -> A-fragment (wave-private region; DS in-order).
#pragma unroll
    for (int nt = 0; nt < 2; ++nt)
#pragma unroll
      for (int r = 0; r < 8; ++r)
        Pb[w][hi * 8 + r][nt * 16 + lm] = f2bf(p[nt][r]);
    v16bf pa = cat16(*(const v8bf*)&Pb[w][lm][hi * 8],
                     *(const v8bf*)&Pb[w][lm][16 + hi * 8]);

#pragma unroll
    for (int j = 0; j < 4; ++j) {
      const int dk = j * 16 + lm;
      v16bf vb = cat16(*(const v8bf*)&Vt[dk][hi * 8],
                       *(const v8bf*)&Vt[dk][16 + hi * 8]);
      acc[j] = __builtin_amdgcn_wmma_f32_16x16x32_bf16(
          false, pa, false, vb, (short)0, acc[j], false, false);
    }
    __syncthreads();
  }

#pragma unroll
  for (int j = 0; j < 4; ++j) {
    const int dk = j * 16 + lm;
#pragma unroll
    for (int r = 0; r < 8; ++r) {
      const int qr = q0 + hi * 8 + r;
      ctx[((size_t)(b * S_ + qr) * H_ + h) * DK_ + dk] = acc[j][r] / lstate[r];
    }
  }
}

// ---------------------------------------------------------------------------
// LayerNorm over (S,D) per batch: pass 1 accumulates sum/sumsq (f32 atomics),
// pass 2 applies the affine normalization (and emits a bf16 copy if needed).
// ---------------------------------------------------------------------------
__launch_bounds__(256)
__global__ void reduce_stats_kernel(const float* __restrict__ x,
                                    const float* __restrict__ y,
                                    float* __restrict__ stats) {
  const int b = blockIdx.y;
  const size_t base = (size_t)b * S_ * D_;
  float s = 0.0f, s2 = 0.0f;
  for (int i = blockIdx.x * blockDim.x + threadIdx.x; i < S_ * D_;
       i += gridDim.x * blockDim.x) {
    float v = x[base + i] + y[base + i];
    s += v; s2 += v * v;
  }
#pragma unroll
  for (int off = 16; off; off >>= 1) {
    s  += __shfl_xor(s,  off, 32);
    s2 += __shfl_xor(s2, off, 32);
  }
  __shared__ float sh[16];
  const int w = threadIdx.x >> 5, lane = threadIdx.x & 31;
  if (lane == 0) { sh[w] = s; sh[w + 8] = s2; }
  __syncthreads();
  if (threadIdx.x == 0) {
    float a = 0.0f, a2 = 0.0f;
    for (int i = 0; i < 8; ++i) { a += sh[i]; a2 += sh[i + 8]; }
    atomicAdd(&stats[b * 2 + 0], a);
    atomicAdd(&stats[b * 2 + 1], a2);
  }
}

template<bool WBF16>
__launch_bounds__(256)
__global__ void ln_apply_kernel(const float* __restrict__ x,
                                const float* __restrict__ y,
                                const float* __restrict__ stats,
                                const float* __restrict__ g,
                                const float* __restrict__ beta,
                                float* __restrict__ outf,
                                __bf16* __restrict__ outb) {
  const int b = blockIdx.y;
  const int i = blockIdx.x * blockDim.x + threadIdx.x;  // 0 .. S*D-1
  const float invN = 1.0f / (float)(S_ * D_);
  const float mu   = stats[b * 2 + 0] * invN;
  const float var  = stats[b * 2 + 1] * invN - mu * mu;
  const float rstd = rsqrtf(var + EPS_);
  const size_t base = (size_t)b * S_ * D_;
  float v = x[base + i] + y[base + i];
  float o = (v - mu) * rstd * g[i] + beta[i];
  outf[base + i] = o;
  if (WBF16) outb[base + i] = f2bf(o);
}

// ---------------------------------------------------------------------------
extern "C" void kernel_launch(void* const* d_in, const int* in_sizes, int n_in,
                              void* d_out, int out_size, void* d_ws, size_t ws_size,
                              hipStream_t stream) {
  (void)in_sizes; (void)n_in; (void)out_size; (void)ws_size;
  const float* x   = (const float*)d_in[0];
  const float* wq  = (const float*)d_in[1];
  const float* bq  = (const float*)d_in[2];
  const float* wk  = (const float*)d_in[3];
  const float* bk  = (const float*)d_in[4];
  const float* wv  = (const float*)d_in[5];
  const float* bv  = (const float*)d_in[6];
  const float* g1  = (const float*)d_in[7];
  const float* be1 = (const float*)d_in[8];
  const float* w1  = (const float*)d_in[9];
  const float* b1  = (const float*)d_in[10];
  const float* w2  = (const float*)d_in[11];
  const float* b2  = (const float*)d_in[12];
  const float* g2  = (const float*)d_in[13];
  const float* be2 = (const float*)d_in[14];
  float* out = (float*)d_out;

  const size_t MSD = (size_t)B_ * S_ * D_;          // 8M
  const size_t MDF = (size_t)B_ * S_ * DFF_;        // 33.5M
  char* p = (char*)d_ws;
  auto bump = [&](size_t bytes) -> char* {
    char* r = p; p += (bytes + 255) & ~(size_t)255; return r;
  };
  __bf16* xb   = (__bf16*)bump(MSD * 2);
  __bf16* wqb  = (__bf16*)bump((size_t)D_ * D_ * 2);
  __bf16* wkb  = (__bf16*)bump((size_t)D_ * D_ * 2);
  __bf16* wvb  = (__bf16*)bump((size_t)D_ * D_ * 2);
  __bf16* w1b  = (__bf16*)bump((size_t)D_ * DFF_ * 2);
  __bf16* w2b  = (__bf16*)bump((size_t)DFF_ * D_ * 2);
  __bf16* qb   = (__bf16*)bump(MSD * 2);
  __bf16* kb   = (__bf16*)bump(MSD * 2);
  __bf16* vb   = (__bf16*)bump(MSD * 2);
  float*  ctx  = (float*)bump(MSD * 4);
  float*  hbuf = (float*)bump(MSD * 4);
  __bf16* hb   = (__bf16*)bump(MSD * 2);
  __bf16* ff1  = (__bf16*)bump(MDF * 2);
  float*  ff2  = (float*)bump(MSD * 4);
  float*  stats = (float*)bump(256);

  const int M = B_ * S_;  // 8192

  zero_stats_kernel<<<1, 64, 0, stream>>>(stats);

  cast_f32_bf16_kernel<<<(int)(MSD / 256), 256, 0, stream>>>(x, xb, (int)MSD);
  cast_f32_bf16_kernel<<<D_ * D_ / 256, 256, 0, stream>>>(wq, wqb, D_ * D_);
  cast_f32_bf16_kernel<<<D_ * D_ / 256, 256, 0, stream>>>(wk, wkb, D_ * D_);
  cast_f32_bf16_kernel<<<D_ * D_ / 256, 256, 0, stream>>>(wv, wvb, D_ * D_);
  cast_f32_bf16_kernel<<<D_ * DFF_ / 256, 256, 0, stream>>>(w1, w1b, D_ * DFF_);
  cast_f32_bf16_kernel<<<DFF_ * D_ / 256, 256, 0, stream>>>(w2, w2b, DFF_ * D_);

  // QKV projections (bf16 outputs; [B,S,D] row-major == [B,S,H,DK] view).
  dim3 gQKV(D_ / 64, M / 64);
  gemm_wmma_kernel<false, false, true><<<gQKV, 128, 0, stream>>>(
      xb, wqb, bq, nullptr, qb, M, D_, D_);
  gemm_wmma_kernel<false, false, true><<<gQKV, 128, 0, stream>>>(
      xb, wkb, bk, nullptr, kb, M, D_, D_);
  gemm_wmma_kernel<false, false, true><<<gQKV, 128, 0, stream>>>(
      xb, wvb, bv, nullptr, vb, M, D_, D_);

  // Attention.
  attn_wmma_kernel<<<dim3(S_ / 64, B_ * H_), 128, 0, stream>>>(qb, kb, vb, ctx);

  // LN1 over (x + ctx).
  reduce_stats_kernel<<<dim3(256, B_), 256, 0, stream>>>(x, ctx, stats);
  ln_apply_kernel<true><<<dim3(S_ * D_ / 256, B_), 256, 0, stream>>>(
      x, ctx, stats, g1, be1, hbuf, hb);

  // FFN.
  gemm_wmma_kernel<true, false, true><<<dim3(DFF_ / 64, M / 64), 128, 0, stream>>>(
      hb, w1b, b1, nullptr, ff1, M, DFF_, D_);
  gemm_wmma_kernel<false, true, false><<<dim3(D_ / 64, M / 64), 128, 0, stream>>>(
      ff1, w2b, b2, ff2, nullptr, M, D_, DFF_);

  // LN2 over (h + ff) -> output.
  reduce_stats_kernel<<<dim3(256, B_), 256, 0, stream>>>(hbuf, ff2, stats + 16);
  ln_apply_kernel<false><<<dim3(S_ * D_ / 256, B_), 256, 0, stream>>>(
      hbuf, ff2, stats + 16, g2, be2, out, nullptr);
}
